// TransformerLayer_57629871177836
// MI455X (gfx1250) — matrix operand order repaired and checked by reading, not compiled, not run on hardware
//
#include <hip/hip_runtime.h>
#include <stdint.h>

// ---------------------------------------------------------------------------
// CDNA5 (gfx1250) transformer layer: bf16 WMMA everywhere, f32 accumulate.
// GEMMs use async global->LDS copies (ASYNCcnt) with LDS double-buffering.
// ---------------------------------------------------------------------------

typedef __attribute__((ext_vector_type(16))) __bf16 v16bf;
typedef __attribute__((ext_vector_type(8)))  float  v8f;

union Frag16 { v16bf v; uint4 q[2]; };

__device__ __forceinline__ uint16_t f2b(float f) {
  union { float f; uint32_t u; } cv; cv.f = f;
  uint32_t u = cv.u;
  return (uint16_t)((u + 0x7FFFu + ((u >> 16) & 1u)) >> 16);  // RNE f32->bf16
}

// Async global->LDS copy, 16B per lane. Generic pointers to __shared__ carry
// the LDS byte offset in their low 32 bits (ISA 10.2).
__device__ __forceinline__ void async_copy_b128(uint32_t lds, const void* g) {
  asm volatile("global_load_async_to_lds_b128 %0, %1, off"
               :: "v"(lds), "v"(g) : "memory");
}
__device__ __forceinline__ void wait_async0() {
  asm volatile("s_wait_asynccnt 0x0" ::: "memory");
}
__device__ __forceinline__ uint32_t lds_addr_of(const void* p) {
  return (uint32_t)(uintptr_t)p;
}

// ---------------------------------------------------------------------------
// fp32 -> bf16 conversion (grid-stride)
// ---------------------------------------------------------------------------
__global__ __launch_bounds__(256) void cvt_f32_bf16(const float* __restrict__ in,
                                                    uint16_t* __restrict__ out, int n) {
  for (int i = blockIdx.x * blockDim.x + threadIdx.x; i < n;
       i += gridDim.x * blockDim.x)
    out[i] = f2b(in[i]);
}

// ---------------------------------------------------------------------------
// Tiled bf16 GEMM: C[M,N] = A[M,K] @ W[N,K]^T (+bias), WMMA f32 accumulate.
// Block tile 128x128, K-step 32, double-buffered async global->LDS staging.
// 256 threads = 8 waves, wave grid 4(M) x 2(N), per-wave 32x64 = 8 C frags.
// omode: 0=f32 out, 1=bf16 out, 2=bf16 out + ReLU,
//        3=bf16 out with [b,h,s,j] head remap (Q/K),
//        4=bf16 out with [b,h,j,s] transposed remap (V^T).
// Requires M%128==0, N%128==0, K%32==0 (true for all calls here).
// ---------------------------------------------------------------------------
#define BM 128
#define BN 128
#define BKP 40   // 32 bf16 + 8 pad (80B row stride, 16B aligned, bank-staggered)

__global__ __launch_bounds__(256) void gemm_bf16(
    const uint16_t* __restrict__ A, const uint16_t* __restrict__ W,
    const float* __restrict__ bias,
    float* __restrict__ outF, uint16_t* __restrict__ outB,
    int M, int N, int K, int omode, int S, int nh, int dh) {
  __shared__ __attribute__((aligned(16))) uint16_t As[2][BM * BKP];
  __shared__ __attribute__((aligned(16))) uint16_t Bs[2][BN * BKP];

  const int tid  = threadIdx.x;
  const int lane = tid & 31;
  const int w    = tid >> 5;       // wave32
  const int waveM = w & 3, waveN = w >> 2;
  const int m0 = blockIdx.y * BM, n0 = blockIdx.x * BN;

  const int lrow = tid >> 1;       // 0..127: row this thread stages
  const int lseg = tid & 1;        // 16-element half of the 32-wide k-slab

  // per-thread global sources and LDS destinations for async staging
  const uint16_t* gA = A + (size_t)(m0 + lrow) * K + lseg * 16;
  const uint16_t* gW = W + (size_t)(n0 + lrow) * K + lseg * 16;
  uint32_t ldsA[2], ldsB[2];
  ldsA[0] = lds_addr_of(&As[0][lrow * BKP + lseg * 16]);
  ldsA[1] = lds_addr_of(&As[1][lrow * BKP + lseg * 16]);
  ldsB[0] = lds_addr_of(&Bs[0][lrow * BKP + lseg * 16]);
  ldsB[1] = lds_addr_of(&Bs[1][lrow * BKP + lseg * 16]);

  v8f acc[8];
#pragma unroll
  for (int f = 0; f < 8; ++f)
#pragma unroll
    for (int i = 0; i < 8; ++i) acc[f][i] = 0.f;

  // prologue: stage k-slab 0 into buffer 0 (4 x 16B per thread)
  async_copy_b128(ldsA[0],      gA);
  async_copy_b128(ldsA[0] + 16, gA + 8);
  async_copy_b128(ldsB[0],      gW);
  async_copy_b128(ldsB[0] + 16, gW + 8);

  const int nk = K >> 5;
  for (int t = 0; t < nk; ++t) {
    const int cur = t & 1;
    wait_async0();        // this wave's copies into buf[cur] are in LDS
    __syncthreads();      // all waves' copies done; all done reading buf[1-cur]

    if (t + 1 < nk) {     // stage next k-slab while computing this one
      const int kk = (t + 1) << 5;
      async_copy_b128(ldsA[1 - cur],      gA + kk);
      async_copy_b128(ldsA[1 - cur] + 16, gA + kk + 8);
      async_copy_b128(ldsB[1 - cur],      gW + kk);
      async_copy_b128(ldsB[1 - cur] + 16, gW + kk + 8);
    }

    // ---- fragments from LDS, 8 WMMAs ----
    Frag16 af[2], bf[4];
#pragma unroll
    for (int mt = 0; mt < 2; ++mt) {
      int base = (waveM * 32 + mt * 16 + (lane & 15)) * BKP + ((lane >> 4) * 8);
      af[mt].q[0] = *(const uint4*)&As[cur][base];        // K = half*8 .. +7
      af[mt].q[1] = *(const uint4*)&As[cur][base + 16];   // K = 16+half*8 .. +7
    }
#pragma unroll
    for (int bt = 0; bt < 4; ++bt) {
      int base = (waveN * 64 + bt * 16 + (lane & 15)) * BKP + ((lane >> 4) * 8);
      bf[bt].q[0] = *(const uint4*)&Bs[cur][base];        // K = half*8 .. +7
      bf[bt].q[1] = *(const uint4*)&Bs[cur][base + 16];   // K = 16+half*8 .. +7
    }
#pragma unroll
    for (int mt = 0; mt < 2; ++mt)
#pragma unroll
      for (int bt = 0; bt < 4; ++bt)
        acc[mt * 4 + bt] = __builtin_amdgcn_wmma_f32_16x16x32_bf16(
            false, af[mt].v, false, bf[bt].v, (short)0, acc[mt * 4 + bt],
            false, false);
  }

  // ---- epilogue: bias (+ReLU), store with layout remap ----
#pragma unroll
  for (int mt = 0; mt < 2; ++mt) {
#pragma unroll
    for (int bt = 0; bt < 4; ++bt) {
      v8f c = acc[mt * 4 + bt];
      int gn = n0 + waveN * 64 + bt * 16 + (lane & 15);
      float bv = bias ? bias[gn] : 0.f;
#pragma unroll
      for (int i = 0; i < 8; ++i) {
        int gm = m0 + waveM * 32 + mt * 16 + i + ((lane >> 4) << 3);
        float v = c[i] + bv;
        if (omode == 2) v = v > 0.f ? v : 0.f;
        if (omode == 0) {
          outF[(size_t)gm * N + gn] = v;
        } else if (omode <= 2) {
          outB[(size_t)gm * N + gn] = f2b(v);
        } else {
          int bb = gm / S, ss = gm - bb * S;
          int hh = gn / dh, jj = gn - hh * dh;
          size_t idx = (omode == 3)
                           ? ((size_t)(bb * nh + hh) * S + ss) * dh + jj
                           : ((size_t)(bb * nh + hh) * dh + jj) * S + ss;
          outB[idx] = f2b(v);
        }
      }
    }
  }
}

// ---------------------------------------------------------------------------
// Flash attention (causal). One WG (4 waves, 128 thr) handles 32 query rows of
// one (b,h). k-blocks of 32 keys: each wave computes one 16x16 score tile
// (16 WMMAs over d=512), cooperative online softmax via LDS, then each wave
// accumulates its 128-wide j-chunk of O (16 WMMA accumulators).
// Q/K: bf16 [b*h, S, dh] row-major;  Vt: bf16 [b*h, dh, S] (transposed).
// O: bf16 [b, s, nh*dh].
// ---------------------------------------------------------------------------
#define AQ 32
#define AKB 32

__global__ __launch_bounds__(128) void attn_flash(
    const uint16_t* __restrict__ Q, const uint16_t* __restrict__ Kk,
    const uint16_t* __restrict__ Vt, uint16_t* __restrict__ O,
    int S, int dh, int nh, float scale) {
  __shared__ float    Sf[32 * 33];     // raw scores, f32
  __shared__ __attribute__((aligned(16))) uint16_t Pb[32 * BKP];
  __shared__ float    mrow[32], lrowS[32], arow[32];

  const int tid = threadIdx.x, lane = tid & 31, w = tid >> 5;
  const int q0 = blockIdx.x * AQ;
  const int bh = blockIdx.y;
  const int b = bh / nh, h = bh - b * nh;
  const uint16_t* Qb = Q  + (size_t)bh * S * dh;
  const uint16_t* Kb = Kk + (size_t)bh * S * dh;
  const uint16_t* Vb = Vt + (size_t)bh * dh * S;

  const int smt = w & 1, snt = w >> 1;  // this wave's score tile
  const int j0 = w * 128;               // this wave's O column chunk

  v8f acc[16];
#pragma unroll
  for (int f = 0; f < 16; ++f)
#pragma unroll
    for (int i = 0; i < 8; ++i) acc[f][i] = 0.f;

  if (tid < 32) { mrow[tid] = -1e30f; lrowS[tid] = 0.f; }
  __syncthreads();

  const int nkb = q0 / AKB;             // causal: k-blocks 0..q0/32
  for (int kb = 0; kb <= nkb; ++kb) {
    // ---- scores tile (smt, snt): Q[16, dh] @ K[16, dh]^T ----
    v8f sc;
#pragma unroll
    for (int i = 0; i < 8; ++i) sc[i] = 0.f;
    for (int kk = 0; kk < dh; kk += 32) {
      Frag16 a, bq;
      int ao = (q0 + smt * 16 + (lane & 15)) * dh + kk + ((lane >> 4) * 8);
      a.q[0] = *(const uint4*)(Qb + ao);
      a.q[1] = *(const uint4*)(Qb + ao + 16);
      int bo2 = (kb * AKB + snt * 16 + (lane & 15)) * dh + kk + ((lane >> 4) * 8);
      bq.q[0] = *(const uint4*)(Kb + bo2);
      bq.q[1] = *(const uint4*)(Kb + bo2 + 16);
      sc = __builtin_amdgcn_wmma_f32_16x16x32_bf16(false, a.v, false, bq.v,
                                                   (short)0, sc, false, false);
    }
#pragma unroll
    for (int i = 0; i < 8; ++i)
      Sf[(smt * 16 + i + ((lane >> 4) << 3)) * 33 + snt * 16 + (lane & 15)] =
          sc[i] * scale;
    __syncthreads();

    // ---- online softmax row pass (32 rows, one thread each) ----
    if (tid < 32) {
      int r = tid, q = q0 + r;
      int kmax = q - kb * AKB + 1;        // causal: k <= q
      if (kmax > AKB) kmax = AKB;
      float mx = -1e30f;
      for (int c = 0; c < kmax; ++c) {
        float s = Sf[r * 33 + c];
        mx = s > mx ? s : mx;
      }
      float mo = mrow[r];
      float mn = mx > mo ? mx : mo;
      float al = __expf(mo - mn);
      float sum = 0.f;
      for (int c = 0; c < AKB; ++c) {
        float ps = (c < kmax) ? __expf(Sf[r * 33 + c] - mn) : 0.f;
        Pb[r * BKP + c] = f2b(ps);
        sum += ps;
      }
      mrow[r] = mn;
      lrowS[r] = lrowS[r] * al + sum;
      arow[r] = al;
    }
    __syncthreads();

    // ---- rescale O accumulators, then O += P @ V ----
#pragma unroll
    for (int mt = 0; mt < 2; ++mt) {
      Frag16 pf;
      int pb = (mt * 16 + (lane & 15)) * BKP + ((lane >> 4) * 8);
      pf.q[0] = *(const uint4*)&Pb[pb];
      pf.q[1] = *(const uint4*)&Pb[pb + 16];
      float av[8];
#pragma unroll
      for (int i = 0; i < 8; ++i)
        av[i] = arow[mt * 16 + i + ((lane >> 4) << 3)];
#pragma unroll
      for (int jt = 0; jt < 8; ++jt) {
        int f = mt * 8 + jt;
#pragma unroll
        for (int i = 0; i < 8; ++i) acc[f][i] *= av[i];
        Frag16 vf;
        int vo = (j0 + jt * 16 + (lane & 15)) * S + kb * AKB + ((lane >> 4) * 8);
        vf.q[0] = *(const uint4*)(Vb + vo);
        vf.q[1] = *(const uint4*)(Vb + vo + 16);
        acc[f] = __builtin_amdgcn_wmma_f32_16x16x32_bf16(
            false, pf.v, false, vf.v, (short)0, acc[f], false, false);
      }
    }
    __syncthreads();
  }

  // ---- finalize: O /= l, store bf16 into [b, s, nh*dh] ----
  const int Dp = nh * dh;
#pragma unroll
  for (int mt = 0; mt < 2; ++mt) {
    float li[8];
#pragma unroll
    for (int i = 0; i < 8; ++i)
      li[i] = 1.f / lrowS[mt * 16 + i + ((lane >> 4) << 3)];
#pragma unroll
    for (int jt = 0; jt < 8; ++jt) {
      int f = mt * 8 + jt;
      int col = j0 + jt * 16 + (lane & 15);
#pragma unroll
      for (int i = 0; i < 8; ++i) {
        int row = q0 + mt * 16 + i + ((lane >> 4) << 3);
        O[(size_t)(b * S + row) * Dp + h * dh + col] = f2b(acc[f][i] * li[i]);
      }
    }
  }
}

// ---------------------------------------------------------------------------
// LayerNorm over d=512 with optional residual; f32 out and/or bf16 out.
// One block (256 thr) per row; each thread owns cols t and t+256.
// ---------------------------------------------------------------------------
__global__ __launch_bounds__(256) void ln_res(
    const float* __restrict__ x, const float* __restrict__ res,
    const float* __restrict__ g, const float* __restrict__ bb,
    float* __restrict__ outF, uint16_t* __restrict__ outB, int d) {
  __shared__ float s1[256], s2[256];
  const int row = blockIdx.x, t = threadIdx.x;
  const float* xr = x + (size_t)row * d;
  float v0 = xr[t]       + (res ? res[(size_t)row * d + t]       : 0.f);
  float v1 = xr[t + 256] + (res ? res[(size_t)row * d + t + 256] : 0.f);
  s1[t] = v0 + v1;
  s2[t] = v0 * v0 + v1 * v1;
  __syncthreads();
  for (int st = 128; st > 0; st >>= 1) {
    if (t < st) { s1[t] += s1[t + st]; s2[t] += s2[t + st]; }
    __syncthreads();
  }
  float mu   = s1[0] / (float)d;
  float var  = s2[0] / (float)d - mu * mu;
  float rinv = rsqrtf(var + 1e-5f);
  float o0 = (v0 - mu) * rinv * g[t] + bb[t];
  float o1 = (v1 - mu) * rinv * g[t + 256] + bb[t + 256];
  if (outF) {
    outF[(size_t)row * d + t]       = o0;
    outF[(size_t)row * d + t + 256] = o1;
  }
  if (outB) {
    outB[(size_t)row * d + t]       = f2b(o0);
    outB[(size_t)row * d + t + 256] = f2b(o1);
  }
}

// ---------------------------------------------------------------------------
// Host-side orchestration
// ---------------------------------------------------------------------------
extern "C" void kernel_launch(void* const* d_in, const int* in_sizes, int n_in,
                              void* d_out, int out_size, void* d_ws, size_t ws_size,
                              hipStream_t stream) {
  (void)in_sizes; (void)n_in; (void)out_size; (void)ws_size;
  const float* x   = (const float*)d_in[0];
  const float* wq  = (const float*)d_in[1];
  const float* bq  = (const float*)d_in[2];
  const float* wk  = (const float*)d_in[3];
  const float* bk  = (const float*)d_in[4];
  const float* wv  = (const float*)d_in[5];
  const float* bv  = (const float*)d_in[6];
  const float* wo  = (const float*)d_in[7];
  const float* bo  = (const float*)d_in[8];
  const float* g1  = (const float*)d_in[9];
  const float* b1  = (const float*)d_in[10];
  const float* wf1 = (const float*)d_in[11];
  const float* bf1 = (const float*)d_in[12];
  const float* wf2 = (const float*)d_in[13];
  const float* bf2 = (const float*)d_in[14];
  const float* g2  = (const float*)d_in[15];
  const float* b2  = (const float*)d_in[16];

  const int B = 8, S = 1024, D = 512, NH = 8, DP = 4096, DF = 2048;
  const int MS = B * S;  // 8192 rows

  // --- carve workspace ---
  char* p = (char*)d_ws;
  auto take = [&](size_t bytes) -> void* {
    void* r = (void*)p;
    p += (bytes + 255) & ~(size_t)255;
    return r;
  };
  uint16_t* xb    = (uint16_t*)take((size_t)MS * D * 2);
  uint16_t* wqb   = (uint16_t*)take((size_t)DP * D * 2);
  uint16_t* wkb   = (uint16_t*)take((size_t)DP * D * 2);
  uint16_t* wvb   = (uint16_t*)take((size_t)DP * D * 2);
  uint16_t* wob   = (uint16_t*)take((size_t)D * DP * 2);
  uint16_t* w1b   = (uint16_t*)take((size_t)DF * D * 2);
  uint16_t* w2b   = (uint16_t*)take((size_t)D * DF * 2);
  uint16_t* Qh    = (uint16_t*)take((size_t)B * NH * S * D * 2);
  uint16_t* Kh    = (uint16_t*)take((size_t)B * NH * S * D * 2);
  uint16_t* Vth   = (uint16_t*)take((size_t)B * NH * D * S * 2);
  uint16_t* Ob    = (uint16_t*)take((size_t)MS * DP * 2);
  float*    aF    = (float*)take((size_t)MS * D * 4);
  float*    abar  = (float*)take((size_t)MS * D * 4);
  uint16_t* abarb = (uint16_t*)take((size_t)MS * D * 2);
  uint16_t* h1    = (uint16_t*)take((size_t)MS * DF * 2);
  float*    h2    = (float*)take((size_t)MS * D * 4);

  auto cvt = [&](const float* in, uint16_t* out, size_t n) {
    int blocks = (int)((n + 255) / 256);
    if (blocks > 4096) blocks = 4096;
    cvt_f32_bf16<<<dim3(blocks), dim3(256), 0, stream>>>(in, out, (int)n);
  };

  // --- 1. fp32 -> bf16 for activations and weights ---
  cvt(x,   xb,  (size_t)MS * D);
  cvt(wq,  wqb, (size_t)DP * D);
  cvt(wk,  wkb, (size_t)DP * D);
  cvt(wv,  wvb, (size_t)DP * D);
  cvt(wo,  wob, (size_t)D * DP);
  cvt(wf1, w1b, (size_t)DF * D);
  cvt(wf2, w2b, (size_t)D * DF);

  // --- 2. QKV projections (head-major outputs; V transposed) ---
  gemm_bf16<<<dim3(DP / BN, MS / BM), 256, 0, stream>>>(
      xb, wqb, bq, nullptr, Qh, MS, DP, D, 3, S, NH, D);
  gemm_bf16<<<dim3(DP / BN, MS / BM), 256, 0, stream>>>(
      xb, wkb, bk, nullptr, Kh, MS, DP, D, 3, S, NH, D);
  gemm_bf16<<<dim3(DP / BN, MS / BM), 256, 0, stream>>>(
      xb, wvb, bv, nullptr, Vth, MS, DP, D, 4, S, NH, D);

  // --- 3. causal flash attention ---
  attn_flash<<<dim3(S / AQ, B * NH), 128, 0, stream>>>(
      Qh, Kh, Vth, Ob, S, D, NH, 0.044194173824159216f);  // 1/sqrt(512)

  // --- 4. output projection (f32 out for residual) ---
  gemm_bf16<<<dim3(D / BN, MS / BM), 256, 0, stream>>>(
      Ob, wob, bo, aF, nullptr, MS, D, DP, 0, S, NH, D);

  // --- 5. LN1(a + x) -> abar (f32) + abarb (bf16) ---
  ln_res<<<dim3(MS), dim3(256), 0, stream>>>(aF, x, g1, b1, abar, abarb, D);

  // --- 6. FFN: h1 = relu(abar @ w_fc1^T + b_fc1) (bf16) ---
  gemm_bf16<<<dim3(DF / BN, MS / BM), 256, 0, stream>>>(
      abarb, w1b, bf1, nullptr, h1, MS, DF, D, 2, S, NH, D);

  // --- 7. h2 = h1 @ w_fc2^T + b_fc2 (f32) ---
  gemm_bf16<<<dim3(D / BN, MS / BM), 256, 0, stream>>>(
      h1, w2b, bf2, h2, nullptr, MS, D, DF, 0, S, NH, D);

  // --- 8. LN2(h2 + abar) -> d_out (f32) ---
  ln_res<<<dim3(MS), dim3(256), 0, stream>>>(h2, abar, g2, b2,
                                             (float*)d_out, nullptr, D);
}
